// MODEL_1_55989193671245
// MI455X (gfx1250) — matrix operand order repaired and checked by reading, not compiled, not run on hardware
//
#include <hip/hip_runtime.h>

typedef __attribute__((ext_vector_type(16))) _Float16 v16h;
typedef __attribute__((ext_vector_type(8)))  _Float16 v8h;
typedef __attribute__((ext_vector_type(8)))  float    v8f;

#define N_NODES 100000
#define E_EDGES 1600000
#define D_FEAT  128
#define L_SEQ   50
#define B_BATCH 1024
#define C_CLS   10

// ---------------------------------------------------------------- zero
__global__ void zero_kernel(float* __restrict__ p, long n) {
  long i = (long)blockIdx.x * blockDim.x + threadIdx.x;
  if (i < n) p[i] = 0.0f;
}

// ---------------------------------------------------------------- WMMA GEMM
// C[M x Nn] = op(A[M x K]) * Bw + bias   (f16 WMMA, f32 accumulate)
//   - A fp32, leading dim lda; VEC=true when (lda%4==0 && base 16B aligned)
//     so A-operand fetch is global_load_b128 instead of 16x b32
//   - optional row gather (readout stage); RELU_IN/RELU_OUT compile-time
//   - Bw is K x Nn row-major, or (transB) Nn x K row-major (weight2.T case)
// Block: 256 threads = 8 waves; wave w covers rows [blk*128 + w*16, +16),
// full Nn width as NT tiles of 16. Weights converted to f16 in LDS once
// per block, stored column-major so B-operand fetch is 2x ds_load_b128.
template<int KK, int NT, bool VEC, bool RELU_IN, bool RELU_OUT>
__global__ __launch_bounds__(256)
void gemm_wmma(const float* __restrict__ A, int lda,
               const float* __restrict__ Bw, const float* __restrict__ bias,
               float* __restrict__ C, int M, int Nn,
               const int* __restrict__ gather, int transB)
{
  __shared__ _Float16 Bt[NT * 16 * KK];   // [col][k], contiguous in k

  for (int i = threadIdx.x; i < NT * 16 * KK; i += 256) {
    int col = i / KK, k = i % KK;
    float v = 0.0f;
    if (col < Nn) v = transB ? Bw[col * KK + k] : Bw[k * Nn + col];
    Bt[i] = (_Float16)v;
  }
  __syncthreads();

  const int wave = threadIdx.x >> 5;
  const int lane = threadIdx.x & 31;
  const int hi   = (lane & 16) ? 1 : 0;
  const int l15  = lane & 15;
  const int r0   = blockIdx.x * 128 + wave * 16;

  // A row owned by this lane (lanes 0-15 and 16-31 both cover M=0..15)
  int rowOut = r0 + l15;
  int rclamp = rowOut < M ? rowOut : (M - 1);
  long rowSrc = gather ? (long)gather[rclamp] : (long)rclamp;
  const float* __restrict__ arow = A + rowSrc * (long)lda;

  v8f acc[NT];
#pragma unroll
  for (int t = 0; t < NT; ++t) acc[t] = (v8f){0, 0, 0, 0, 0, 0, 0, 0};

#pragma unroll 1
  for (int k0 = 0; k0 < KK; k0 += 32) {
    // A operand: ISA 16-bit A 16x32 layout.
    // lanes 0-15: K = k0+[0..7] and k0+[16..23]; lanes 16-31: +8 shifted.
    const int kb = k0 + (hi ? 8 : 0);
    float av[16];
    if (VEC) {
      float4 f0 = *(const float4*)(arow + kb);
      float4 f1 = *(const float4*)(arow + kb + 4);
      float4 f2 = *(const float4*)(arow + kb + 16);
      float4 f3 = *(const float4*)(arow + kb + 20);
      av[0]=f0.x; av[1]=f0.y; av[2]=f0.z;  av[3]=f0.w;
      av[4]=f1.x; av[5]=f1.y; av[6]=f1.z;  av[7]=f1.w;
      av[8]=f2.x; av[9]=f2.y; av[10]=f2.z; av[11]=f2.w;
      av[12]=f3.x;av[13]=f3.y;av[14]=f3.z; av[15]=f3.w;
    } else {
#pragma unroll
      for (int i = 0; i < 8; ++i) {
        av[i]     = arow[kb + i];
        av[8 + i] = arow[kb + 16 + i];
      }
    }
    v16h a;
#pragma unroll
    for (int i = 0; i < 16; ++i) {
      float v0 = av[i];
      if (RELU_IN) v0 = v0 > 0.f ? v0 : 0.f;
      a[i] = (_Float16)v0;
    }
    // B operand: lanes 0-15 hold K=k0+[0..7] and k0+[16..23] of col l15, lanes 16-31: +8 shifted
    const int koff = k0 + (hi ? 8 : 0);
#pragma unroll
    for (int t = 0; t < NT; ++t) {
      const int col = t * 16 + l15;
      const _Float16* bp = &Bt[col * KK + koff];
      v8h b0 = *(const v8h*)(bp);
      v8h b1 = *(const v8h*)(bp + 16);
      v16h b = __builtin_shufflevector(b0, b1, 0, 1, 2, 3, 4, 5, 6, 7,
                                               8, 9, 10, 11, 12, 13, 14, 15);
      acc[t] = __builtin_amdgcn_wmma_f32_16x16x32_f16(
          /*neg_a=*/false, a, /*neg_b=*/false, b,
          /*c_mod=*/(short)0, acc[t], /*reuse_a=*/false, /*reuse_b=*/false);
    }
  }

  // C/D layout: element i -> row r0 + i + 8*hi, col = t*16 + l15
#pragma unroll
  for (int t = 0; t < NT; ++t) {
    const int col = t * 16 + l15;
    const float bv = (col < Nn) ? bias[col] : 0.0f;
#pragma unroll
    for (int i = 0; i < 8; ++i) {
      const int row = r0 + i + hi * 8;
      if (row < M && col < Nn) {
        float v = acc[t][i] + bv;
        if (RELU_OUT) v = v > 0.f ? v : 0.f;
        C[(long)row * Nn + col] = v;
      }
    }
  }
}

// ---------------------------------------------------------------- SpMM
// Y[r,:] += v * X[c,:]  via segment-sum atomics; one wave per edge,
// float4 per lane (32 lanes x 4 = 128 features).
__global__ __launch_bounds__(256)
void spmm_kernel(const int* __restrict__ rows, const int* __restrict__ cols,
                 const float* __restrict__ vals, const float* __restrict__ X,
                 float* __restrict__ Y, int E)
{
  int e = blockIdx.x * 8 + (threadIdx.x >> 5);
  if (e >= E) return;
  int lane = threadIdx.x & 31;
  int c = cols[e];
  int r = rows[e];
  float v = vals[e];
  const float4 xv = *(const float4*)(X + (long)c * D_FEAT + lane * 4);
  float* yp = Y + (long)r * D_FEAT + lane * 4;
  atomicAdd(yp + 0, v * xv.x);
  atomicAdd(yp + 1, v * xv.y);
  atomicAdd(yp + 2, v * xv.z);
  atomicAdd(yp + 3, v * xv.w);
}

// ---------------------------------------------------------------- combine
// x1 = a0 * n  +  s * (1 - n)      (n broadcast along feature dim)
__global__ void combine_kernel(const float* __restrict__ a0,
                               const float* __restrict__ s,
                               const float* __restrict__ n,
                               float* __restrict__ x1, long total)
{
  long i = (long)blockIdx.x * blockDim.x + threadIdx.x;
  if (i < total) {
    float np = n[i >> 7];
    x1[i] = a0[i] * np + s[i] * (1.0f - np);
  }
}

// ---------------------------------------------------------------- head
// per batch element: einsum('bcl,c->bl') for both branches, concat,
// classifier matmul + argmax. out[0:B] = argmax (as float), out[B:] = sel.
__global__ __launch_bounds__(64)
void head_kernel(const float* __restrict__ s1m, const float* __restrict__ s2m,
                 const float* __restrict__ w1, const float* __restrict__ b1,
                 const float* __restrict__ w2, const float* __restrict__ b2,
                 const float* __restrict__ Wc, const float* __restrict__ bc,
                 float* __restrict__ out)
{
  int b = blockIdx.x * blockDim.x + threadIdx.x;
  if (b >= B_BATCH) return;

  float sel[2 * L_SEQ];
  // s2 branch first (sel[0:50])
  for (int l = 0; l < L_SEQ; ++l) {
    float acc = 0.0f;
    for (int c = 0; c < L_SEQ; ++c)
      acc += s2m[((long)b * L_SEQ + c) * L_SEQ + l] * w2[c];
    sel[l] = acc + b2[0];
  }
  // s1 branch (sel[50:100])
  for (int l = 0; l < L_SEQ; ++l) {
    float acc = 0.0f;
    for (int c = 0; c < L_SEQ; ++c)
      acc += s1m[((long)b * L_SEQ + c) * L_SEQ + l] * w1[c];
    sel[L_SEQ + l] = acc + b1[0];
  }

  float* o = out + B_BATCH + (long)b * (2 * L_SEQ);
  for (int j = 0; j < 2 * L_SEQ; ++j) o[j] = sel[j];

  int best = 0;
  float bv = -3.0e38f;
  for (int c = 0; c < C_CLS; ++c) {
    float p = bc[c];
    for (int j = 0; j < 2 * L_SEQ; ++j) p += sel[j] * Wc[j * C_CLS + c];
    if (p > bv) { bv = p; best = c; }
  }
  out[b] = (float)best;
}

// ---------------------------------------------------------------- launch
extern "C" void kernel_launch(void* const* d_in, const int* in_sizes, int n_in,
                              void* d_out, int out_size, void* d_ws, size_t ws_size,
                              hipStream_t stream)
{
  const float* A1       = (const float*)d_in[0];   // N x 257
  const int*   adj_rows = (const int*)  d_in[1];   // E
  const int*   adj_cols = (const int*)  d_in[2];   // E
  const float* adj_vals = (const float*)d_in[3];   // E
  const int*   batch_idx= (const int*)  d_in[4];   // B x 1 x L
  const float* Lin1     = (const float*)d_in[5];   // 256 x 128
  const float* Lin1_b   = (const float*)d_in[6];   // 128
  const float* n_param  = (const float*)d_in[7];   // N x 1
  const float* gc1_w    = (const float*)d_in[8];   // 256 x 128
  const float* gc1_b    = (const float*)d_in[9];
  const float* gc2_w    = (const float*)d_in[10];  // 128 x 128
  const float* gc2_b    = (const float*)d_in[11];
  const float* gc3_w    = (const float*)d_in[12];  // 128 x 128
  const float* gc3_b    = (const float*)d_in[13];
  const float* weight2  = (const float*)d_in[14];  // 50 x 128 (use as B^T)
  const float* bias2    = (const float*)d_in[15];  // 50
  const float* conv1_w  = (const float*)d_in[16];
  const float* conv1_b  = (const float*)d_in[17];
  const float* conv2_w  = (const float*)d_in[18];
  const float* conv2_b  = (const float*)d_in[19];
  const float* cls_w    = (const float*)d_in[20];  // 100 x 10
  const float* cls_b    = (const float*)d_in[21];  // 10
  float* out = (float*)d_out;

  const long NF = (long)N_NODES * D_FEAT;        // 12.8M floats
  const long SF = (long)B_BATCH * L_SEQ * L_SEQ; // 2.56M floats

  float* a0   = (float*)d_ws;        // keep live for s2 branch
  float* bufA = a0 + NF;
  float* bufB = bufA + NF;
  float* s1m  = bufB + NF;
  float* s2m  = s1m + SF;

  const int GN  = (N_NODES + 127) / 128;          // 782
  const int GS  = (B_BATCH * L_SEQ + 127) / 128;  // 400
  const int GZ  = (int)((NF + 255) / 256);
  const int GE  = (E_EDGES + 7) / 8;
  const int MRO = B_BATCH * L_SEQ;

  // a0 = x @ Lin1 + b        (x = A1[:,1:], lda = 257 -> unaligned, scalar A path)
  gemm_wmma<256, 8, false, false, false><<<GN, 256, 0, stream>>>(
      A1 + 1, 257, Lin1, Lin1_b, a0, N_NODES, D_FEAT, nullptr, 0);
  // t = x @ gc1_w + b
  gemm_wmma<256, 8, false, false, false><<<GN, 256, 0, stream>>>(
      A1 + 1, 257, gc1_w, gc1_b, bufA, N_NODES, D_FEAT, nullptr, 0);
  // bufB = spmm(t)
  zero_kernel<<<GZ, 256, 0, stream>>>(bufB, NF);
  spmm_kernel<<<GE, 256, 0, stream>>>(adj_rows, adj_cols, adj_vals, bufA, bufB, E_EDGES);
  // bufA = x1 = a0*n + bufB*(1-n)
  combine_kernel<<<GZ, 256, 0, stream>>>(a0, bufB, n_param, bufA, NF);
  // bufB = x1 @ gc2_w + b ; bufA = spmm(bufB)
  gemm_wmma<128, 8, true, false, false><<<GN, 256, 0, stream>>>(
      bufA, 128, gc2_w, gc2_b, bufB, N_NODES, D_FEAT, nullptr, 0);
  zero_kernel<<<GZ, 256, 0, stream>>>(bufA, NF);
  spmm_kernel<<<GE, 256, 0, stream>>>(adj_rows, adj_cols, adj_vals, bufB, bufA, E_EDGES);
  // bufB = _ @ gc3_w + b ; bufA = x2 = spmm(bufB)
  gemm_wmma<128, 8, true, false, false><<<GN, 256, 0, stream>>>(
      bufA, 128, gc3_w, gc3_b, bufB, N_NODES, D_FEAT, nullptr, 0);
  zero_kernel<<<GZ, 256, 0, stream>>>(bufA, NF);
  spmm_kernel<<<GE, 256, 0, stream>>>(adj_rows, adj_cols, adj_vals, bufB, bufA, E_EDGES);

  // s1 = gather(x2, idx) @ weight2.T + bias2        (M=51200, K=128, N=50)
  gemm_wmma<128, 4, true, false, false><<<GS, 256, 0, stream>>>(
      bufA, 128, weight2, bias2, s1m, MRO, L_SEQ, batch_idx, /*transB=*/1);
  // s2 = relu(relu(gather(a0, idx)) @ weight2.T + bias2)
  gemm_wmma<128, 4, true, true, true><<<GS, 256, 0, stream>>>(
      a0, 128, weight2, bias2, s2m, MRO, L_SEQ, batch_idx, /*transB=*/1);

  // head: conv reductions + concat + classifier + argmax
  head_kernel<<<B_BATCH / 64, 64, 0, stream>>>(s1m, s2m, conv1_w, conv1_b,
      conv2_w, conv2_b, cls_w, cls_b, out);
}
